// Basic_CrossAD_37649683317445
// MI455X (gfx1250) — matrix operand order repaired and checked by reading, not compiled, not run on hardware
//
#include <hip/hip_runtime.h>

// ---------------------------------------------------------------------------
// Types for CDNA5 WMMA (wave32, bf16 -> f32 accumulate)
// ---------------------------------------------------------------------------
typedef __bf16 v16bf __attribute__((ext_vector_type(16)));
typedef __bf16 v8bf  __attribute__((ext_vector_type(8)));
typedef float  v8f   __attribute__((ext_vector_type(8)));

// ---------------------------------------------------------------------------
// Problem constants (from reference)
// ---------------------------------------------------------------------------
constexpr int BS      = 8;
constexpr int CH      = 32;
constexpr int BROWS   = BS * CH;      // 256 independent series
constexpr int SEQ     = 2048;
constexpr int PATCH   = 16;
constexpr int DMODEL  = 512;
constexpr int NHEADS  = 8;
constexpr int HD      = 64;
constexpr int DFF     = 2048;
constexpr int LTOT    = 16 + 32 + 64 + 128;   // 240 tokens
constexpr int MTOK    = BROWS * LTOT;         // 61440 rows

// ---------------------------------------------------------------------------
// Weight convert: f32 (K x N) -> bf16 transposed (N x K) so that B-matrix
// WMMA fragments are contiguous in memory (N-major, K contiguous).
// ---------------------------------------------------------------------------
__global__ void cvt_transpose_kernel(const float* __restrict__ src,
                                     __bf16* __restrict__ dst, int K, int N) {
  int idx = blockIdx.x * blockDim.x + threadIdx.x;
  if (idx >= K * N) return;
  int k = idx / N;
  int n = idx - k * N;
  dst[(size_t)n * K + k] = (__bf16)src[idx];
}

// ---------------------------------------------------------------------------
// Embedding: h[b,l,:] = sum_j x[b, k*(p*16+j)] * W_emb[j,:] + pos[l,:]
// grid = (MTOK, 2), block = 256  (d = blockIdx.y*256 + tid)
// ---------------------------------------------------------------------------
__global__ void embed_kernel(const float* __restrict__ x_enc,
                             const float* __restrict__ w_emb,
                             float* __restrict__ h32,
                             __bf16* __restrict__ hb) {
  int m = blockIdx.x;              // b*240 + l
  int b = m / LTOT;
  int l = m - b * LTOT;
  int d = blockIdx.y * 256 + threadIdx.x;

  // segment decode
  int kstep, p;
  if      (l < 16)  { kstep = 8; p = l;       }
  else if (l < 48)  { kstep = 4; p = l - 16;  }
  else if (l < 112) { kstep = 2; p = l - 48;  }
  else              { kstep = 1; p = l - 112; }

  __shared__ float patch[PATCH];
  if (threadIdx.x < PATCH) {
    int t_idx = kstep * (p * PATCH + threadIdx.x);
    patch[threadIdx.x] = x_enc[((size_t)(b >> 5) * SEQ + t_idx) * CH + (b & 31)];
  }
  __syncthreads();

  float acc = 0.f;
#pragma unroll
  for (int j = 0; j < PATCH; ++j)
    acc += patch[j] * w_emb[j * DMODEL + d];

  // sinusoidal positional embedding: pe row r = p * kstep
  float r   = (float)(p * kstep);
  float div = __expf(-__logf(10000.0f) * (float)(d & ~1) / (float)DMODEL);
  float pos = (d & 1) ? __cosf(r * div) : __sinf(r * div);
  acc += pos;

  size_t idx = (size_t)m * DMODEL + d;
  h32[idx] = acc;
  hb[idx]  = (__bf16)acc;
}

// ---------------------------------------------------------------------------
// WMMA GEMM:  C(MxN) = A(MxK, bf16 row-major) * Bt^T + bias, with Bt (NxK).
// K and N are compile-time so all tile offsets fold into the 24-bit immediate
// offset of global_load_b128: the K-loop uses exactly two base address
// registers (A, B) and no per-iteration VALU -> no WMMA->VALU hazard NOPs,
// and the scheduler can software-pipeline loads across WMMAs.
// Block = 256 threads (8 waves). Each wave computes a 32x64 strip:
//   2 M-subtiles x 4 N-subtiles = 8 x v_wmma_f32_16x16x32_bf16 per K-step,
//   fed by 4 A-loads + 8 B-loads (1.5 b128 loads / WMMA).
// Grid = (M/256, N/64). Requires M%256==0 (holds: MTOK=61440).
// ---------------------------------------------------------------------------
template <bool GELU, bool WF32, bool WB16, int K, int N>
__global__ void gemm_bf16_wmma(const __bf16* __restrict__ A,
                               const __bf16* __restrict__ Bt,
                               const float* __restrict__ bias,
                               float* __restrict__ Cf,
                               __bf16* __restrict__ Cb) {
  const int wave = threadIdx.x >> 5;
  const int lane = threadIdx.x & 31;
  const int hi   = lane >> 4;
  const int nl   = lane & 15;

  const int row0 = blockIdx.x * 256 + wave * 32;
  const int col0 = blockIdx.y * 64;

  // A base: ISA A layout — lanes 0-15 / 16-31 hold the same 16 M rows but
  // different K halves (elements 0..7 -> K=8*hi+e, 8..15 -> K=16+8*hi+e-8).
  // Second M-subtile is a constant +16*K element offset (immediate).
  const __bf16* aBase = A + (size_t)(row0 + nl) * K + 8 * hi;
  // B base: lane nl selects output column; same K layout as A (0..7 -> K=8*hi+e, 8..15 -> K=16+8*hi+e-8).
  // N-subtile nt is a constant +nt*16*K element offset (immediate).
  const __bf16* bBase = Bt + (size_t)(col0 + nl) * K + 8 * hi;

  v8f acc[2][4] = {};

#pragma unroll 16
  for (int k0 = 0; k0 < K; k0 += 32) {
    v8bf a00 = *(const v8bf*)(aBase + k0);
    v8bf a01 = *(const v8bf*)(aBase + k0 + 16);
    v8bf a10 = *(const v8bf*)(aBase + 16 * K + k0);
    v8bf a11 = *(const v8bf*)(aBase + 16 * K + k0 + 16);
    v16bf af0 = __builtin_shufflevector(a00, a01, 0, 1, 2, 3, 4, 5, 6, 7,
                                        8, 9, 10, 11, 12, 13, 14, 15);
    v16bf af1 = __builtin_shufflevector(a10, a11, 0, 1, 2, 3, 4, 5, 6, 7,
                                        8, 9, 10, 11, 12, 13, 14, 15);
#pragma unroll
    for (int nt = 0; nt < 4; ++nt) {
      v8bf b0 = *(const v8bf*)(bBase + nt * 16 * K + k0);
      v8bf b1 = *(const v8bf*)(bBase + nt * 16 * K + k0 + 16);
      v16bf bf = __builtin_shufflevector(b0, b1, 0, 1, 2, 3, 4, 5, 6, 7,
                                         8, 9, 10, 11, 12, 13, 14, 15);
      acc[0][nt] = __builtin_amdgcn_wmma_f32_16x16x32_bf16(
          false, af0, false, bf, (short)0, acc[0][nt], false, false);
      acc[1][nt] = __builtin_amdgcn_wmma_f32_16x16x32_bf16(
          false, af1, false, bf, (short)0, acc[1][nt], false, false);
    }
  }

  // Epilogue: lane holds (m = row0 + mt*16 + 8*hi + r, n = col0 + nt*16 + nl)
#pragma unroll
  for (int mt = 0; mt < 2; ++mt) {
#pragma unroll
    for (int nt = 0; nt < 4; ++nt) {
      int n = col0 + nt * 16 + nl;
      float bn = bias[n];
#pragma unroll
      for (int r = 0; r < 8; ++r) {
        float v = acc[mt][nt][r] + bn;
        if (GELU) {
          float x = v;
          v = 0.5f * x *
              (1.f + tanhf(0.7978845608028654f * (x + 0.044715f * x * x * x)));
        }
        size_t idx = (size_t)(row0 + mt * 16 + hi * 8 + r) * N + n;
        if (WF32) Cf[idx] = v;
        if (WB16) Cb[idx] = (__bf16)v;
      }
    }
  }
}

// ---------------------------------------------------------------------------
// Block-diagonal attention. One block per (batch row, head, segment).
// Segments: {len,off} = {16,0},{32,16},{64,48},{128,112}. 8 waves/block,
// each wave owns query rows l = wave, wave+8, ... (uniform trip count since
// all segment lengths are multiples of 8). Softmax via wave shfl reductions.
// Q rows are held in registers as 8 x b128 bf16 chunks; K rows streamed as
// b128 chunks.
// ---------------------------------------------------------------------------
__global__ void attention_kernel(const __bf16* __restrict__ q,
                                 const __bf16* __restrict__ k,
                                 const __bf16* __restrict__ v,
                                 __bf16* __restrict__ out) {
  const int bid  = blockIdx.x;          // b*32 + head*4 + seg
  const int b    = bid >> 5;
  const int rem  = bid & 31;
  const int head = rem >> 2;
  const int seg  = rem & 3;

  const int lsTab[4]  = {16, 32, 64, 128};
  const int offTab[4] = {0, 16, 48, 112};
  const int Ls    = lsTab[seg];
  const int mbase = b * LTOT + offTab[seg];
  const int dh0   = head * HD;
  const float scale = 0.125f;           // 1/sqrt(64)

  const int wave = threadIdx.x >> 5;
  const int lane = threadIdx.x & 31;

  __shared__ float probs[8][128];

  const int iters = Ls / 8;
  for (int it = 0; it < iters; ++it) {
    const int l = wave + it * 8;

    // Q row in registers (64 bf16 = 8 x v8bf)
    const v8bf* qp = (const v8bf*)(q + (size_t)(mbase + l) * DMODEL + dh0);
    v8bf qv[8];
#pragma unroll
    for (int c = 0; c < 8; ++c) qv[c] = qp[c];

    float sc[4];
#pragma unroll
    for (int jj = 0; jj < 4; ++jj) {
      int key = lane + 32 * jj;
      float s = -1e30f;
      if (key < Ls) {
        s = 0.f;
        const v8bf* kp = (const v8bf*)(k + (size_t)(mbase + key) * DMODEL + dh0);
#pragma unroll
        for (int c = 0; c < 8; ++c) {
          v8bf kc = kp[c];
#pragma unroll
          for (int e = 0; e < 8; ++e)
            s += (float)qv[c][e] * (float)kc[e];
        }
        s *= scale;
      }
      sc[jj] = s;
    }

    float mx = fmaxf(fmaxf(sc[0], sc[1]), fmaxf(sc[2], sc[3]));
#pragma unroll
    for (int off = 16; off > 0; off >>= 1)
      mx = fmaxf(mx, __shfl_xor(mx, off, 32));

    float p[4];
    float ssum = 0.f;
#pragma unroll
    for (int jj = 0; jj < 4; ++jj) {
      int key = lane + 32 * jj;
      p[jj] = (key < Ls) ? __expf(sc[jj] - mx) : 0.f;
      ssum += p[jj];
    }
#pragma unroll
    for (int off = 16; off > 0; off >>= 1)
      ssum += __shfl_xor(ssum, off, 32);
    float inv = 1.f / ssum;

#pragma unroll
    for (int jj = 0; jj < 4; ++jj) {
      int key = lane + 32 * jj;
      if (key < Ls) probs[wave][key] = p[jj] * inv;
    }
    __syncthreads();

    for (int d2 = lane; d2 < HD; d2 += 32) {
      float o = 0.f;
      for (int j = 0; j < Ls; ++j)
        o += probs[wave][j] * (float)v[(size_t)(mbase + j) * DMODEL + dh0 + d2];
      out[(size_t)(mbase + l) * DMODEL + dh0 + d2] = (__bf16)o;
    }
    __syncthreads();
  }
}

// ---------------------------------------------------------------------------
// Residual + LayerNorm over D=512. Block per token, 256 threads, 2 elems each.
// outF may alias x (in-place safe: reads happen before writes).
// ---------------------------------------------------------------------------
__global__ void res_ln_kernel(const float* __restrict__ x,
                              const float* __restrict__ add,
                              const float* __restrict__ g,
                              const float* __restrict__ bsh,
                              float* __restrict__ outF,
                              __bf16* __restrict__ outB) {
  const int m = blockIdx.x;
  const int t = threadIdx.x;
  const size_t i0 = (size_t)m * DMODEL + t;
  const size_t i1 = i0 + 256;

  float v0 = x[i0] + (add ? add[i0] : 0.f);
  float v1 = x[i1] + (add ? add[i1] : 0.f);

  __shared__ float red[256];
  red[t] = v0 + v1;
  __syncthreads();
#pragma unroll
  for (int s = 128; s > 0; s >>= 1) {
    if (t < s) red[t] += red[t + s];
    __syncthreads();
  }
  float mu = red[0] * (1.f / DMODEL);
  __syncthreads();

  float d0 = v0 - mu, d1 = v1 - mu;
  red[t] = d0 * d0 + d1 * d1;
  __syncthreads();
#pragma unroll
  for (int s = 128; s > 0; s >>= 1) {
    if (t < s) red[t] += red[t + s];
    __syncthreads();
  }
  float rstd = rsqrtf(red[0] * (1.f / DMODEL) + 1e-5f);

  float y0 = d0 * rstd * g[t] + bsh[t];
  float y1 = d1 * rstd * g[t + 256] + bsh[t + 256];
  outF[i0] = y0;
  outF[i1] = y1;
  if (outB) { outB[i0] = (__bf16)y0; outB[i1] = (__bf16)y1; }
}

// ---------------------------------------------------------------------------
// Host-side orchestration
// ---------------------------------------------------------------------------
extern "C" void kernel_launch(void* const* d_in, const int* in_sizes, int n_in,
                              void* d_out, int out_size, void* d_ws, size_t ws_size,
                              hipStream_t stream) {
  (void)in_sizes; (void)n_in; (void)out_size; (void)ws_size;

  const float* x_enc = (const float*)d_in[0];
  const float* W_emb = (const float*)d_in[1];
  const float* Wq    = (const float*)d_in[2];
  const float* bq    = (const float*)d_in[3];
  const float* Wk    = (const float*)d_in[4];
  const float* bk    = (const float*)d_in[5];
  const float* Wv    = (const float*)d_in[6];
  const float* bv    = (const float*)d_in[7];
  const float* Wo    = (const float*)d_in[8];
  const float* bo    = (const float*)d_in[9];
  const float* ln1g  = (const float*)d_in[10];
  const float* ln1b  = (const float*)d_in[11];
  const float* W1    = (const float*)d_in[12];
  const float* b1    = (const float*)d_in[13];
  const float* W2    = (const float*)d_in[14];
  const float* b2    = (const float*)d_in[15];
  const float* ln2g  = (const float*)d_in[16];
  const float* ln2b  = (const float*)d_in[17];
  const float* lnfg  = (const float*)d_in[18];
  const float* lnfb  = (const float*)d_in[19];
  float* out = (float*)d_out;

  // ---- workspace arena -------------------------------------------------
  char* ws = (char*)d_ws;
  const size_t SZ_F32 = (size_t)MTOK * DMODEL * 4;   // 125,829,120
  const size_t SZ_B16 = (size_t)MTOK * DMODEL * 2;   //  62,914,560
  float*  h32 = (float*)(ws);
  float*  o32 = (float*)(ws + SZ_F32);
  __bf16* hb  = (__bf16*)(ws + 2 * SZ_F32);
  __bf16* qb  = (__bf16*)(ws + 2 * SZ_F32 + 1 * SZ_B16);
  __bf16* kb  = (__bf16*)(ws + 2 * SZ_F32 + 2 * SZ_B16);
  __bf16* vb  = (__bf16*)(ws + 2 * SZ_F32 + 3 * SZ_B16);
  __bf16* ab  = (__bf16*)(ws + 2 * SZ_F32 + 4 * SZ_B16);
  __bf16* mid = qb;  // FFN mid (MTOK x DFF bf16) reuses q/k/v/a region exactly
  char*   wbase = ws + 2 * SZ_F32 + 5 * SZ_B16;

  const size_t W_DD  = (size_t)DMODEL * DMODEL * 2;  // 524,288 bytes
  const size_t W_DF  = (size_t)DMODEL * DFF    * 2;  // 2,097,152 bytes
  const size_t W_LYR = 4 * W_DD + 2 * W_DF;          // per-layer stride
  auto wqT = [&](int L) { return (__bf16*)(wbase + L * W_LYR); };
  auto wkT = [&](int L) { return (__bf16*)(wbase + L * W_LYR + 1 * W_DD); };
  auto wvT = [&](int L) { return (__bf16*)(wbase + L * W_LYR + 2 * W_DD); };
  auto woT = [&](int L) { return (__bf16*)(wbase + L * W_LYR + 3 * W_DD); };
  auto w1T = [&](int L) { return (__bf16*)(wbase + L * W_LYR + 4 * W_DD); };
  auto w2T = [&](int L) { return (__bf16*)(wbase + L * W_LYR + 4 * W_DD + W_DF); };

  // ---- weight conversion (f32 KxN -> bf16 NxK) -------------------------
  const int gDD = (DMODEL * DMODEL + 255) / 256;
  const int gDF = (DMODEL * DFF + 255) / 256;
  for (int L = 0; L < 3; ++L) {
    const size_t oDD = (size_t)L * DMODEL * DMODEL;
    const size_t oDF = (size_t)L * DMODEL * DFF;
    cvt_transpose_kernel<<<gDD, 256, 0, stream>>>(Wq + oDD, wqT(L), DMODEL, DMODEL);
    cvt_transpose_kernel<<<gDD, 256, 0, stream>>>(Wk + oDD, wkT(L), DMODEL, DMODEL);
    cvt_transpose_kernel<<<gDD, 256, 0, stream>>>(Wv + oDD, wvT(L), DMODEL, DMODEL);
    cvt_transpose_kernel<<<gDD, 256, 0, stream>>>(Wo + oDD, woT(L), DMODEL, DMODEL);
    cvt_transpose_kernel<<<gDF, 256, 0, stream>>>(W1 + oDF, w1T(L), DMODEL, DFF);
    cvt_transpose_kernel<<<gDF, 256, 0, stream>>>(W2 + oDF, w2T(L), DFF, DMODEL);
  }

  // ---- embedding -------------------------------------------------------
  embed_kernel<<<dim3(MTOK, DMODEL / 256), 256, 0, stream>>>(x_enc, W_emb, h32, hb);

  // ---- encoder layers --------------------------------------------------
  const dim3 gProj(MTOK / 256, DMODEL / 64);   // 240 x 8
  const dim3 gFF1 (MTOK / 256, DFF / 64);      // 240 x 32
  for (int L = 0; L < 3; ++L) {
    // Q, K, V projections (bf16 out for attention)
    gemm_bf16_wmma<false, false, true, DMODEL, DMODEL><<<gProj, 256, 0, stream>>>(
        hb, wqT(L), bq + L * DMODEL, nullptr, qb);
    gemm_bf16_wmma<false, false, true, DMODEL, DMODEL><<<gProj, 256, 0, stream>>>(
        hb, wkT(L), bk + L * DMODEL, nullptr, kb);
    gemm_bf16_wmma<false, false, true, DMODEL, DMODEL><<<gProj, 256, 0, stream>>>(
        hb, wvT(L), bv + L * DMODEL, nullptr, vb);

    // block-diagonal attention per (batch, head, segment)
    attention_kernel<<<BROWS * NHEADS * 4, 256, 0, stream>>>(qb, kb, vb, ab);

    // output projection (f32 out for residual)
    gemm_bf16_wmma<false, true, false, DMODEL, DMODEL><<<gProj, 256, 0, stream>>>(
        ab, woT(L), bo + L * DMODEL, o32, nullptr);

    // h = LN(h + attn_out)
    res_ln_kernel<<<MTOK, 256, 0, stream>>>(h32, o32, ln1g + L * DMODEL,
                                            ln1b + L * DMODEL, h32, hb);

    // FFN: mid = gelu(h @ W1 + b1) [bf16 only]; f = mid @ W2 + b2 [f32]
    gemm_bf16_wmma<true, false, true, DMODEL, DFF><<<gFF1, 256, 0, stream>>>(
        hb, w1T(L), b1 + L * DFF, nullptr, mid);
    gemm_bf16_wmma<false, true, false, DFF, DMODEL><<<gProj, 256, 0, stream>>>(
        mid, w2T(L), b2 + L * DMODEL, o32, nullptr);

    // h = LN(h + f)
    res_ln_kernel<<<MTOK, 256, 0, stream>>>(h32, o32, ln2g + L * DMODEL,
                                            ln2b + L * DMODEL, h32, hb);
  }

  // ---- final layer norm -> output -------------------------------------
  res_ln_kernel<<<MTOK, 256, 0, stream>>>(h32, nullptr, lnfg, lnfb, out, nullptr);
}
